// TRMAttention_89378269430415
// MI455X (gfx1250) — matrix operand order repaired and checked by reading, not compiled, not run on hardware
//
#include <hip/hip_runtime.h>
#include <hip/hip_bf16.h>
#include <math.h>

// ---------------------------------------------------------------------------
// MI455X (gfx1250) wave32 WMMA attention:
//   Q/K/V = X @ W{q,k,v}^T  -> RoPE(Q,K) -> causal flash attention -> @ Wo^T
// bf16 data path (v_wmma_f32_16x16x32_bf16), fp32 accumulation + softmax.
// Async global->LDS copies (GLOBAL_LOAD_ASYNC_TO_LDS_B128, ASYNCcnt).
// ---------------------------------------------------------------------------

typedef __attribute__((ext_vector_type(16))) __bf16 v16bf;
typedef __attribute__((ext_vector_type(8)))  float  v8f;

#define HIDDEN   2048
#define HEADS    16
#define HEAD_DIM 128
#define BATCH    2
#define SEQ      2048
#define M_TOTAL  (BATCH * SEQ)      // 4096 rows of X

__device__ __forceinline__ v8f vzero8() {
    v8f z;
#pragma unroll
    for (int i = 0; i < 8; ++i) z[i] = 0.0f;
    return z;
}

__device__ __forceinline__ v8f wmma_bf16(v16bf a, v16bf b, v8f c) {
    // (neg_a, A, neg_b, B, c_mod, C, reuse_a, reuse_b)
    return __builtin_amdgcn_wmma_f32_16x16x32_bf16(false, a, false, b,
                                                   (short)0, c, false, false);
}

// Async 16-byte global -> LDS copy (per lane).  LDS address = low 32 bits of
// the generic shared pointer (flat->LDS truncates to addr[31:0]).
__device__ __forceinline__ void async_cp16(void* lds, const void* g) {
    unsigned lds_off = (unsigned)(uintptr_t)lds;
    asm volatile("global_load_async_to_lds_b128 %0, %1, off"
                 :: "v"(lds_off), "v"(g)
                 : "memory");
}
__device__ __forceinline__ void async_wait0() {
    asm volatile("s_wait_asynccnt 0" ::: "memory");
}

// A-fragment (16x32, MxK) for bf16: lanes 0-15 hold rows M=0..15, lanes 16-31
// the same rows' other K-halves.  Per lane: two contiguous 16-byte runs:
//   k in [half*8, half*8+7]  and  k in [16+half*8, 16+half*8+7]
__device__ __forceinline__ v16bf load_a_frag(const __bf16* base, int ld, int lane) {
    const int m    = lane & 15;
    const int half = lane >> 4;
    union { v16bf v; uint4 q[2]; } u;
    const __bf16* r = base + (size_t)m * ld + half * 8;
    u.q[0] = *reinterpret_cast<const uint4*>(r);
    u.q[1] = *reinterpret_cast<const uint4*>(r + 16);
    return u.v;
}

// B-fragment (32x16, KxN) for bf16 where the source is stored "N rows, K
// contiguous" (true for W[n][k], K[n][d], transposed-V).  Lane holds column
// n = lane&15; two 16-byte runs: k in [half*8, half*8+7] and [16+half*8, 16+half*8+7].
__device__ __forceinline__ v16bf load_b_frag(const __bf16* base, int ld, int lane) {
    const int n  = lane & 15;
    const int kb = (lane >> 4) * 8;
    union { v16bf v; uint4 q[2]; } u;
    const __bf16* r = base + (size_t)n * ld + kb;
    u.q[0] = *reinterpret_cast<const uint4*>(r);
    u.q[1] = *reinterpret_cast<const uint4*>(r + 16);
    return u.v;
}

// ---------------------------------------------------------------------------
// fp32 -> bf16 conversion (bandwidth-trivial; working set fits in 192MB L2)
// ---------------------------------------------------------------------------
__global__ void cvt_f32_bf16(const float* __restrict__ in,
                             __bf16* __restrict__ out, int n) {
    int i = blockIdx.x * blockDim.x + threadIdx.x;
    int stride = gridDim.x * blockDim.x;
    for (; i < n; i += stride) out[i] = (__bf16)in[i];
}

// ---------------------------------------------------------------------------
// WMMA GEMM:  Y[m,n] = sum_k X[m,k] * W[n,k]   (nn.Linear: x @ W^T)
// Block tile 128x128, K-step 64, 256 threads = 8 waves (2x4), each wave 64x32
// -> 16 WMMAs per wave per barrier.  Double-buffered LDS with async copies.
// OUTMODE 0: bf16 into [B,H,S,HEAD_DIM] headed layout (Q/K/V)
// OUTMODE 1: fp32 into [M, HIDDEN] flat layout (final output)
// ---------------------------------------------------------------------------
template <int OUTMODE>
__global__ __launch_bounds__(256)
void gemm_wmma(const __bf16* __restrict__ X, const __bf16* __restrict__ W,
               void* __restrict__ out) {
    __shared__ __align__(16) __bf16 Xs[2][128 * 64];   // 2 x 16 KB
    __shared__ __align__(16) __bf16 Ws[2][128 * 64];   // 2 x 16 KB

    const int tid  = threadIdx.x;
    const int lane = tid & 31;
    const int wid  = tid >> 5;
    const int wm   = wid & 1;           // 0..1  (64-row strip)
    const int wn   = wid >> 1;          // 0..3  (32-col strip)
    const int m0   = blockIdx.y * 128;
    const int n0   = blockIdx.x * 128;
    const int half = lane >> 4;
    const int nlo  = lane & 15;

    v8f c[4][2];
#pragma unroll
    for (int i = 0; i < 4; ++i)
#pragma unroll
        for (int j = 0; j < 2; ++j) c[i][j] = vzero8();

    // issue async loads for one 128x64 X tile + 128x64 W tile into buffer buf
    auto issue_tile = [&](int k0, int buf) {
        // 1024 uint4 each, 256 threads -> 4 per thread per tile
#pragma unroll
        for (int v = 0; v < 4; ++v) {
            int j   = tid + v * 256;
            int row = j >> 3, q = j & 7;     // 8 uint4 per 64-elem row
            async_cp16(&Xs[buf][j * 8],
                       X + (size_t)(m0 + row) * HIDDEN + k0 + q * 8);
            async_cp16(&Ws[buf][j * 8],
                       W + (size_t)(n0 + row) * HIDDEN + k0 + q * 8);
        }
    };

    issue_tile(0, 0);

    const int T = HIDDEN / 64;
    for (int t = 0; t < T; ++t) {
        async_wait0();
        __syncthreads();
        if (t + 1 < T) issue_tile((t + 1) * 64, (t + 1) & 1);

        const __bf16* Xb = Xs[t & 1];
        const __bf16* Wb = Ws[t & 1];
#pragma unroll
        for (int kk = 0; kk < 64; kk += 32) {
            v16bf a[4], b[2];
#pragma unroll
            for (int i = 0; i < 4; ++i)
                a[i] = load_a_frag(Xb + (wm * 64 + i * 16) * 64 + kk, 64, lane);
#pragma unroll
            for (int j = 0; j < 2; ++j)
                b[j] = load_b_frag(Wb + (wn * 32 + j * 16) * 64 + kk, 64, lane);
#pragma unroll
            for (int i = 0; i < 4; ++i)
#pragma unroll
                for (int j = 0; j < 2; ++j)
                    c[i][j] = wmma_bf16(a[i], b[j], c[i][j]);
        }
    }

    // C/D layout: VGPR r holds row (r + 8*half), column nlo of each 16x16 tile
#pragma unroll
    for (int i = 0; i < 4; ++i)
#pragma unroll
        for (int j = 0; j < 2; ++j)
#pragma unroll
            for (int r = 0; r < 8; ++r) {
                int m = m0 + wm * 64 + i * 16 + r + 8 * half;
                int n = n0 + wn * 32 + j * 16 + nlo;
                float v = c[i][j][r];
                if (OUTMODE == 0) {
                    int b = m >> 11, s = m & 2047;
                    int h = n >> 7,  d = n & 127;
                    ((__bf16*)out)[(((size_t)(b * HEADS + h)) * SEQ + s) *
                                       HEAD_DIM + d] = (__bf16)v;
                } else {
                    ((float*)out)[(size_t)m * HIDDEN + n] = v;
                }
            }
}

// ---------------------------------------------------------------------------
// RoPE in place on Q and K ([B,H,S,128] bf16).  Pair (i, i+64), i in [0,64).
// ---------------------------------------------------------------------------
__global__ __launch_bounds__(64)
void rope_kernel(__bf16* __restrict__ Q, __bf16* __restrict__ K) {
    const int bhs = blockIdx.x;                 // 0 .. B*H*S-1
    const int s   = bhs & (SEQ - 1);
    const int i   = threadIdx.x;                // 0..63
    const size_t base = (size_t)bhs * HEAD_DIM;

    // inv_freq[i] = theta^(-i/64); ln(10000) = 9.210340371976184
    float inv = __expf(-(float)i * (9.210340371976184f / 64.0f));
    float ang = (float)s * inv;
    float sn, cs;
    __sincosf(ang, &sn, &cs);

    {
        float x1 = (float)Q[base + i], x2 = (float)Q[base + i + 64];
        Q[base + i]      = (__bf16)(x1 * cs - x2 * sn);
        Q[base + i + 64] = (__bf16)(x2 * cs + x1 * sn);
    }
    {
        float x1 = (float)K[base + i], x2 = (float)K[base + i + 64];
        K[base + i]      = (__bf16)(x1 * cs - x2 * sn);
        K[base + i + 64] = (__bf16)(x2 * cs + x1 * sn);
    }
}

// ---------------------------------------------------------------------------
// Causal flash attention.  Grid: (S/64, B*H).  128 threads = 4 waves; each
// wave owns 16 query rows; block owns 64.  KV streamed in 64-wide tiles.
// K tile fill is async (no VGPR round-trip); V is VGPR-transposed into LDS.
// Writes merged [B, S, H*128] bf16 for the output projection.
// ---------------------------------------------------------------------------
__global__ __launch_bounds__(128)
void attn_kernel(const __bf16* __restrict__ Q, const __bf16* __restrict__ K,
                 const __bf16* __restrict__ V, __bf16* __restrict__ AO) {
    __shared__ __align__(16) __bf16 Ks[64 * 128];    // 16 KB
    __shared__ __align__(16) __bf16 Vts[128 * 64];   // 16 KB (transposed V)
    __shared__ __align__(16) __bf16 Ps[4 * 16 * 64]; // 8 KB (per-wave P strip)

    const int tid  = threadIdx.x;
    const int lane = tid & 31;
    const int wid  = tid >> 5;
    const int half = lane >> 4;
    const int nlo  = lane & 15;

    const int bh = blockIdx.y;
    const int b  = bh >> 4, h = bh & 15;
    const int q0 = blockIdx.x * 64;
    const int qr = q0 + wid * 16;                    // this wave's first q row

    const __bf16* Qp = Q + (size_t)bh * SEQ * HEAD_DIM;
    const __bf16* Kp = K + (size_t)bh * SEQ * HEAD_DIM;
    const __bf16* Vp = V + (size_t)bh * SEQ * HEAD_DIM;

    // Preload Q fragments (16 rows x 128 dims = 4 A-frags), held in VGPRs.
    v16bf qf[4];
#pragma unroll
    for (int c4 = 0; c4 < 4; ++c4)
        qf[c4] = load_a_frag(Qp + (size_t)qr * HEAD_DIM + c4 * 32, HEAD_DIM, lane);

    v8f oacc[8];
#pragma unroll
    for (int n2 = 0; n2 < 8; ++n2) oacc[n2] = vzero8();
    float mrow[8], lrow[8];
#pragma unroll
    for (int r = 0; r < 8; ++r) { mrow[r] = -INFINITY; lrow[r] = 0.0f; }

    const float scale = 0.08838834764831845f;        // 1/sqrt(128)
    const int ntiles = (q0 >> 6) + 1;                // causal: tiles up to diag

    for (int t = 0; t < ntiles; ++t) {
        const int kv0 = t * 64;

        // K tile 64x128: async global->LDS, 1024 x 16B / 128 thr = 8 each
        {
            const uint4* kg =
                reinterpret_cast<const uint4*>(Kp + (size_t)kv0 * HEAD_DIM);
#pragma unroll
            for (int j = tid; j < 1024; j += 128)
                async_cp16(&Ks[j * 8], kg + j);
        }
        // V tile transposed into Vts[d][kv] (per-element VGPR transpose)
        for (int j = tid; j < 8192; j += 128) {
            int kv = j >> 7, d = j & 127;
            Vts[d * 64 + kv] = Vp[(size_t)(kv0 + kv) * HEAD_DIM + d];
        }
        async_wait0();
        __syncthreads();

        // S = Q K^T for this wave: 4 column chunks x 4 K-dim chunks
        v8f sc[4];
#pragma unroll
        for (int n = 0; n < 4; ++n) {
            v8f acc = vzero8();
#pragma unroll
            for (int c4 = 0; c4 < 4; ++c4) {
                v16bf bb = load_b_frag(Ks + (size_t)(n * 16) * 128 + c4 * 32,
                                       128, lane);
                acc = wmma_bf16(qf[c4], bb, acc);
            }
            sc[n] = acc;
        }

        // scale + causal mask (diagonal tile only) + per-row tile max
        const bool diag = (t == ntiles - 1);
        float tmax[8];
#pragma unroll
        for (int r = 0; r < 8; ++r) tmax[r] = -INFINITY;
#pragma unroll
        for (int n = 0; n < 4; ++n)
#pragma unroll
            for (int r = 0; r < 8; ++r) {
                float v = sc[n][r] * scale;
                int col = kv0 + n * 16 + nlo;
                int row = qr + r + 8 * half;
                if (diag && col > row) v = -INFINITY;
                sc[n][r] = v;
                tmax[r] = fmaxf(tmax[r], v);
            }
        // row reduction across the 16-lane half-groups (C layout)
#pragma unroll
        for (int msk = 8; msk >= 1; msk >>= 1)
#pragma unroll
            for (int r = 0; r < 8; ++r)
                tmax[r] = fmaxf(tmax[r], __shfl_xor(tmax[r], msk, 32));

        // online softmax rescale
        float rs[8];
#pragma unroll
        for (int r = 0; r < 8; ++r) {
            float nm = fmaxf(mrow[r], tmax[r]);
            float sc_ = __expf(mrow[r] - nm);
            mrow[r] = nm;
            lrow[r] *= sc_;
#pragma unroll
            for (int n2 = 0; n2 < 8; ++n2) oacc[n2][r] *= sc_;
            rs[r] = 0.0f;
        }

        // P = exp(S - m); stash per-wave in LDS (A-frag layout source)
        __bf16* Pw = Ps + wid * (16 * 64);
#pragma unroll
        for (int n = 0; n < 4; ++n)
#pragma unroll
            for (int r = 0; r < 8; ++r) {
                float p = __expf(sc[n][r] - mrow[r]);
                rs[r] += p;
                Pw[(r + 8 * half) * 64 + n * 16 + nlo] = (__bf16)p;
            }
#pragma unroll
        for (int msk = 8; msk >= 1; msk >>= 1)
#pragma unroll
            for (int r = 0; r < 8; ++r) rs[r] += __shfl_xor(rs[r], msk, 32);
#pragma unroll
        for (int r = 0; r < 8; ++r) lrow[r] += rs[r];

        // O += P @ V  (8 d-chunks x 2 kv-chunks)
#pragma unroll
        for (int n2 = 0; n2 < 8; ++n2)
#pragma unroll
            for (int c4 = 0; c4 < 2; ++c4) {
                v16bf pa = load_a_frag(Pw + c4 * 32, 64, lane);
                v16bf vb = load_b_frag(Vts + (size_t)(n2 * 16) * 64 + c4 * 32,
                                       64, lane);
                oacc[n2] = wmma_bf16(pa, vb, oacc[n2]);
            }
        __syncthreads();
    }

    // epilogue: O / l, store merged [B, S, H*128]
#pragma unroll
    for (int n2 = 0; n2 < 8; ++n2)
#pragma unroll
        for (int r = 0; r < 8; ++r) {
            int s = qr + r + 8 * half;
            int d = n2 * 16 + nlo;
            AO[((size_t)b * SEQ + s) * HIDDEN + h * HEAD_DIM + d] =
                (__bf16)(oacc[n2][r] / lrow[r]);
        }
}

// ---------------------------------------------------------------------------
// Host-side orchestration (graph-capture safe: only kernel launches on stream)
// ---------------------------------------------------------------------------
extern "C" void kernel_launch(void* const* d_in, const int* in_sizes, int n_in,
                              void* d_out, int out_size, void* d_ws,
                              size_t ws_size, hipStream_t stream) {
    (void)in_sizes; (void)n_in; (void)out_size; (void)ws_size;

    const float* hs = (const float*)d_in[0];
    const float* wq = (const float*)d_in[1];
    const float* wk = (const float*)d_in[2];
    const float* wv = (const float*)d_in[3];
    const float* wo = (const float*)d_in[4];

    char* ws = (char*)d_ws;
    const size_t NX  = (size_t)M_TOTAL * HIDDEN;     // 8,388,608
    const size_t NW  = (size_t)HIDDEN * HIDDEN;      // 4,194,304
    const size_t NQ  = (size_t)BATCH * HEADS * SEQ * HEAD_DIM; // 8,388,608

    __bf16* Xb  = (__bf16*)(ws);
    __bf16* Wqb = (__bf16*)(ws + NX * 2);
    __bf16* Wkb = (__bf16*)(ws + NX * 2 + NW * 2);
    __bf16* Wvb = (__bf16*)(ws + NX * 2 + NW * 4);
    __bf16* Wob = (__bf16*)(ws + NX * 2 + NW * 6);
    __bf16* Qh  = (__bf16*)(ws + NX * 2 + NW * 8);
    __bf16* Kh  = (__bf16*)(ws + NX * 2 + NW * 8 + NQ * 2);
    __bf16* Vh  = (__bf16*)(ws + NX * 2 + NW * 8 + NQ * 4);
    __bf16* AO  = (__bf16*)(ws + NX * 2 + NW * 8 + NQ * 6);

    // 1) fp32 -> bf16
    cvt_f32_bf16<<<4096, 256, 0, stream>>>(hs, Xb, (int)NX);
    cvt_f32_bf16<<<4096, 256, 0, stream>>>(wq, Wqb, (int)NW);
    cvt_f32_bf16<<<4096, 256, 0, stream>>>(wk, Wkb, (int)NW);
    cvt_f32_bf16<<<4096, 256, 0, stream>>>(wv, Wvb, (int)NW);
    cvt_f32_bf16<<<4096, 256, 0, stream>>>(wo, Wob, (int)NW);

    // 2) Q/K/V projections (WMMA), headed bf16 layout
    dim3 gg(HIDDEN / 128, M_TOTAL / 128);
    gemm_wmma<0><<<gg, 256, 0, stream>>>(Xb, Wqb, (void*)Qh);
    gemm_wmma<0><<<gg, 256, 0, stream>>>(Xb, Wkb, (void*)Kh);
    gemm_wmma<0><<<gg, 256, 0, stream>>>(Xb, Wvb, (void*)Vh);

    // 3) RoPE on Q, K
    rope_kernel<<<BATCH * HEADS * SEQ, 64, 0, stream>>>(Qh, Kh);

    // 4) causal flash attention -> merged [B,S,HIDDEN] bf16
    dim3 ga(SEQ / 64, BATCH * HEADS);
    attn_kernel<<<ga, 128, 0, stream>>>(Qh, Kh, Vh, AO);

    // 5) output projection -> fp32 d_out
    gemm_wmma<1><<<gg, 256, 0, stream>>>(AO, Wob, d_out);
}
